// Grid_Pool_35442070126956
// MI455X (gfx1250) — matrix operand order repaired and checked by reading, not compiled, not run on hardware
//
#include <hip/hip_runtime.h>
#include <hip/hip_bf16.h>

typedef __attribute__((ext_vector_type(16))) _Float16 v16h;
typedef __attribute__((ext_vector_type(8)))  float    v8f;

constexpr int C_IN  = 96;
constexpr int C_OUT = 192;
constexpr int NT    = C_OUT / 16;        // 12 column tiles
constexpr int KT    = C_IN / 32;         // 3 K-steps per tile
constexpr int TPW   = NT / 4;            // 3 column tiles per wave (4 waves/block)
constexpr int HASH_CAP = 1 << 20;        // >2x max unique voxels
constexpr unsigned long long EMPTY_KEY = ~0ull;
constexpr float EPS = 1e-6f;

// ---- order-preserving float <-> uint encoding for atomic max ----
__device__ __forceinline__ unsigned encf(float f) {
    unsigned u = __float_as_uint(f);
    return (u & 0x80000000u) ? ~u : (u | 0x80000000u);
}
__device__ __forceinline__ float decf(unsigned u) {
    unsigned b = (u & 0x80000000u) ? (u & 0x7fffffffu) : ~u;
    return __uint_as_float(b);
}

// ---------------------------------------------------------------- init
__global__ void k_init(unsigned* __restrict__ featU, size_t nfeat,
                       unsigned long long* __restrict__ hkeys,
                       float* __restrict__ counts, float* __restrict__ bsum,
                       float* __restrict__ csum,   float* __restrict__ xsum,
                       int n, int* __restrict__ dimsCtr, size_t tot) {
    size_t stride = (size_t)gridDim.x * blockDim.x;
    for (size_t i = (size_t)blockIdx.x * blockDim.x + threadIdx.x; i < tot; i += stride) {
        if (i < nfeat) featU[i] = 0u;                       // encodes "-inf"
        if (i < (size_t)HASH_CAP) hkeys[i] = EMPTY_KEY;
        if (i < (size_t)n) {
            counts[i] = 0.f; bsum[i] = 0.f; csum[i] = 0.f;
            xsum[3*i] = 0.f; xsum[3*i+1] = 0.f; xsum[3*i+2] = 0.f;
        }
        if (i < 4) dimsCtr[i] = 0;
    }
}

// ---------------------------------------------------------------- dims = max(vox)+1
__global__ void k_dims(const float* __restrict__ xyz, int n, int* __restrict__ dims) {
    int i = blockIdx.x * blockDim.x + threadIdx.x;
    if (i >= n) return;
    #pragma unroll
    for (int a = 0; a < 3; ++a) {
        int v = (int)floorf(xyz[3*i + a] * 0.5f);   // STRIDE = 2
        atomicMax(&dims[a], v + 1);
    }
}

// ---------------------------------------------------------------- W f32 -> f16 in WMMA B-fragment order
// B is 32x16 (KxN) per step: lane = half*16 + n; lanes0-15 hold K={0..7}+{16..23}, lanes16-31 K={8..15}+{24..31},
// element e maps to K = kt*32 + (e<8 ? half*8 + e : 16 + half*8 + e-8).
__global__ void k_wconv(const float* __restrict__ W, _Float16* __restrict__ Wh) {
    int i = blockIdx.x * blockDim.x + threadIdx.x;
    if (i >= NT * KT * 32 * 16) return;
    int e    = i & 15;
    int lane = (i >> 4) & 31;
    int t3   = i >> 9;
    int kt   = t3 % KT;
    int ct   = t3 / KT;
    int nCol = lane & 15, half = lane >> 4;
    int K    = kt * 32 + (e & 8) * 2 + half * 8 + (e & 7);
    Wh[i] = (_Float16)W[(ct * 16 + nCol) * C_IN + K];
}

// ---------------------------------------------------------------- voxel hash insert (claim segment ids)
__global__ void k_hash_insert(const float* __restrict__ xyz, const int* __restrict__ batch,
                              int n, const int* __restrict__ dims,
                              unsigned long long* __restrict__ hkeys, int* __restrict__ hids,
                              int* __restrict__ counter, int* __restrict__ slot) {
    int i = blockIdx.x * blockDim.x + threadIdx.x;
    if (i >= n) return;
    int d0 = dims[0], d1 = dims[1], d2 = dims[2];
    int v0 = (int)floorf(xyz[3*i+0] * 0.5f);
    int v1 = (int)floorf(xyz[3*i+1] * 0.5f);
    int v2 = (int)floorf(xyz[3*i+2] * 0.5f);
    unsigned long long k =
        (unsigned long long)(((((long long)batch[i] * d0 + v0) * d1 + v1) * d2) + v2);
    unsigned pos = (unsigned)((k * 0x9E3779B97F4A7C15ull) >> 40) & (HASH_CAP - 1);
    for (;;) {
        unsigned long long cur = hkeys[pos];
        if (cur == k) break;
        if (cur == EMPTY_KEY) {
            unsigned long long old = atomicCAS(&hkeys[pos], EMPTY_KEY, k);
            if (old == EMPTY_KEY) { hids[pos] = atomicAdd(counter, 1); break; }
            if (old == k) break;
        }
        pos = (pos + 1) & (HASH_CAP - 1);
    }
    slot[i] = (int)pos;
}

// ---------------------------------------------------------------- resolve inv + segment sums
__global__ void k_assign(const float* __restrict__ xyz, const float* __restrict__ xyz_count,
                         const int* __restrict__ batch, int n,
                         const int* __restrict__ hids, const int* __restrict__ slot,
                         int* __restrict__ inv, float* __restrict__ counts,
                         float* __restrict__ bsum, float* __restrict__ csum,
                         float* __restrict__ xsum) {
    int i = blockIdx.x * blockDim.x + threadIdx.x;
    if (i >= n) return;
    int s = hids[slot[i]];
    inv[i] = s;
    float xc = xyz_count[i];
    atomicAdd(&counts[s], 1.f);
    atomicAdd(&bsum[s], (float)batch[i]);
    atomicAdd(&csum[s], xc);
    atomicAdd(&xsum[3*s+0], xyz[3*i+0] * xc);
    atomicAdd(&xsum[3*s+1], xyz[3*i+1] * xc);
    atomicAdd(&xsum[3*s+2], xyz[3*i+2] * xc);
}

// ---------------------------------------------------------------- fused LayerNorm + GEMM (WMMA) + segment-max scatter
__global__ void __launch_bounds__(128)
k_ln_gemm_scatter(const float* __restrict__ feat, const float* __restrict__ gamma,
                  const float* __restrict__ beta, const _Float16* __restrict__ Wh,
                  const int* __restrict__ inv, int n, unsigned* __restrict__ featU) {
    __shared__ float sF[16][C_IN];
    __shared__ __align__(32) _Float16 sA[KT][32][16];   // A fragments, ISA 16-bit layout
    __shared__ int sInv[16];

    const int t    = threadIdx.x;
    const int row0 = blockIdx.x * 16;

    // stage 16x96 f32 tile (coalesced) + the 16 segment ids for this tile
    if (t < 16) sInv[t] = (row0 + t < n) ? inv[row0 + t] : 0;
    for (int i = t; i < 16 * C_IN; i += 128) {
        int r = i / C_IN, c = i - r * C_IN;
        int g = row0 + r;
        sF[r][c] = (g < n) ? feat[(size_t)g * C_IN + c] : 0.f;
    }
    __syncthreads();

    // LayerNorm: 8 lanes per row, 12 channels each; reduce with shfl_xor within 8-lane groups
    {
        int r = t >> 3, l8 = t & 7;
        float s = 0.f, ss = 0.f;
        #pragma unroll
        for (int j = 0; j < 12; ++j) {
            float v = sF[r][l8 * 12 + j];
            s += v; ss += v * v;
        }
        #pragma unroll
        for (int m = 1; m < 8; m <<= 1) {
            s  += __shfl_xor(s,  m, 8);
            ss += __shfl_xor(ss, m, 8);
        }
        float mu   = s * (1.f / C_IN);
        float var  = ss * (1.f / C_IN) - mu * mu;
        float rstd = rsqrtf(var + EPS);
        // normalize -> f16 straight into WMMA A-fragment layout:
        // lanes0-15: row m, K in {0..7}+{16..23}; lanes16-31: row m, K in {8..15}+{24..31}
        #pragma unroll
        for (int j = 0; j < 12; ++j) {
            int ch = l8 * 12 + j;
            float v = (sF[r][ch] - mu) * rstd * gamma[ch] + beta[ch];
            int kt   = ch >> 5;
            int rr   = ch & 31;
            int hi   = (rr >= 16) ? 1 : 0;   // VGPR 4..7 block
            int rm   = rr - hi * 16;         // 0..15
            int half = rm >> 3;              // lane half
            int r2   = rm & 7;
            int e    = (hi * 4 + (r2 >> 1)) * 2 + (r2 & 1);
            sA[kt][half * 16 + r][e] = (_Float16)v;
        }
    }
    __syncthreads();

    const int wave = t >> 5;     // 4 waves, contiguous tile ranges
    const int lane = t & 31;
    const int half = lane >> 4, nIdx = lane & 15;
    const int ct0  = wave * TPW;

    // ---- pre-load ALL fragments so the 6+18 B128 loads overlap, then 9 back-to-back WMMAs
    v16h afrag[KT];
    #pragma unroll
    for (int kt = 0; kt < KT; ++kt)
        afrag[kt] = *(const v16h*)(&sA[kt][lane][0]);

    v16h bfrag[TPW][KT];
    #pragma unroll
    for (int j = 0; j < TPW; ++j)
        #pragma unroll
        for (int kt = 0; kt < KT; ++kt)
            bfrag[j][kt] = *(const v16h*)(Wh + (((size_t)(ct0 + j) * KT + kt) * 32 + lane) * 16);

    v8f acc[TPW];
    #pragma unroll
    for (int j = 0; j < TPW; ++j) acc[j] = v8f{};
    #pragma unroll
    for (int j = 0; j < TPW; ++j)
        #pragma unroll
        for (int kt = 0; kt < KT; ++kt)
            acc[j] = __builtin_amdgcn_wmma_f32_16x16x32_f16(
                false, afrag[kt], false, bfrag[j][kt], (short)0, acc[j], false, false);

    // ---- scatter: segment-max via order-preserving u32 atomics
    // D layout: lane half*16+n holds rows M = v + 8*half in acc[j][v]
    int seg[8];
    #pragma unroll
    for (int v = 0; v < 8; ++v) seg[v] = sInv[v + half * 8];

    if (row0 + 16 <= n) {                         // full tile: uniform EXEC, no branches
        #pragma unroll
        for (int j = 0; j < TPW; ++j) {
            int col = (ct0 + j) * 16 + nIdx;
            #pragma unroll
            for (int v = 0; v < 8; ++v)
                atomicMax(&featU[(size_t)seg[v] * C_OUT + col], encf(acc[j][v]));
        }
    } else {                                      // ragged last tile
        #pragma unroll
        for (int j = 0; j < TPW; ++j) {
            int col = (ct0 + j) * 16 + nIdx;
            #pragma unroll
            for (int v = 0; v < 8; ++v) {
                int g = row0 + v + half * 8;
                if (g < n)
                    atomicMax(&featU[(size_t)seg[v] * C_OUT + col], encf(acc[j][v]));
            }
        }
    }
}

// ---------------------------------------------------------------- finalize outputs
__global__ void k_finalize(const float* __restrict__ counts, const float* __restrict__ bsum,
                           const float* __restrict__ csum, const float* __restrict__ xsum,
                           int n, float* __restrict__ out) {
    unsigned* featU  = (unsigned*)out;                         // aliases feat region (in-place decode)
    float*    oFeat  = out;
    int*      oCoord = (int*)(out + (size_t)n * C_OUT);
    float*    oXyz   = out + (size_t)n * (C_OUT + 4);
    float*    oCnt   = oXyz + (size_t)n * 3;
    int*      oBatch = (int*)(oCnt + n);
    int*      oValid = oBatch + n;

    size_t tot = (size_t)n * C_OUT;
    size_t stride = (size_t)gridDim.x * blockDim.x;
    for (size_t i = (size_t)blockIdx.x * blockDim.x + threadIdx.x; i < tot; i += stride) {
        // decode segment-max features; empty segments -> 0 (reference masks -inf padding)
        unsigned u = featU[i];
        int s = (int)(i / C_OUT);
        oFeat[i] = (counts[s] > 0.f) ? decf(u) : 0.f;

        if (i < (size_t)n) {
            float c    = counts[i];
            float cnt  = fmaxf(csum[i], 1.f);                  // cnt_next = clip(sum, 1)
            float x0 = xsum[3*i+0] / cnt;
            float x1 = xsum[3*i+1] / cnt;
            float x2 = xsum[3*i+2] / cnt;
            int   bn = (int)(bsum[i] / fmaxf(c, 1.f));
            oXyz[3*i+0] = x0; oXyz[3*i+1] = x1; oXyz[3*i+2] = x2;
            oCnt[i]   = cnt;
            oBatch[i] = bn;
            oValid[i] = (c > 0.f) ? 1 : 0;
            oCoord[4*i+0] = bn;
            oCoord[4*i+1] = (int)floorf(x0 * 0.5f);
            oCoord[4*i+2] = (int)floorf(x1 * 0.5f);
            oCoord[4*i+3] = (int)floorf(x2 * 0.5f);
        }
    }
}

// ================================================================ launch
extern "C" void kernel_launch(void* const* d_in, const int* in_sizes, int n_in,
                              void* d_out, int out_size, void* d_ws, size_t ws_size,
                              hipStream_t stream) {
    const float* feat      = (const float*)d_in[0];
    const float* xyz       = (const float*)d_in[1];
    const float* xyz_count = (const float*)d_in[2];
    const int*   batch     = (const int*)d_in[3];
    const float* W         = (const float*)d_in[4];
    const float* ln_gamma  = (const float*)d_in[5];
    const float* ln_beta   = (const float*)d_in[6];
    float* out = (float*)d_out;

    const int n = in_sizes[0] / C_IN;

    // ---- workspace carve-up (256B aligned) ----
    auto align256 = [](size_t x) { return (x + 255) & ~(size_t)255; };
    char* ws = (char*)d_ws;
    size_t off = 0;
    _Float16* Wh = (_Float16*)(ws + off);               off = align256(off + (size_t)C_OUT * C_IN * 2);
    int* dimsCtr = (int*)(ws + off);                    off = align256(off + 4 * sizeof(int));
    unsigned long long* hkeys = (unsigned long long*)(ws + off); off = align256(off + (size_t)HASH_CAP * 8);
    int* hids  = (int*)(ws + off);                      off = align256(off + (size_t)HASH_CAP * 4);
    int* slot  = (int*)(ws + off);                      off = align256(off + (size_t)n * 4);
    int* inv   = (int*)(ws + off);                      off = align256(off + (size_t)n * 4);
    float* counts = (float*)(ws + off);                 off = align256(off + (size_t)n * 4);
    float* bsum   = (float*)(ws + off);                 off = align256(off + (size_t)n * 4);
    float* csum   = (float*)(ws + off);                 off = align256(off + (size_t)n * 4);
    float* xsum   = (float*)(ws + off);                 off = align256(off + (size_t)n * 12);

    unsigned* featU = (unsigned*)out;                    // feat region, encoded during scatter
    size_t nfeat = (size_t)n * C_OUT;

    const int TPB = 256;
    size_t tot_init = nfeat > (size_t)HASH_CAP ? nfeat : (size_t)HASH_CAP;

    k_init<<<4096, TPB, 0, stream>>>(featU, nfeat, hkeys, counts, bsum, csum, xsum,
                                     n, dimsCtr, tot_init);
    k_dims<<<(n + TPB - 1) / TPB, TPB, 0, stream>>>(xyz, n, dimsCtr);
    k_wconv<<<(NT * KT * 32 * 16 + TPB - 1) / TPB, TPB, 0, stream>>>(W, Wh);
    k_hash_insert<<<(n + TPB - 1) / TPB, TPB, 0, stream>>>(xyz, batch, n, dimsCtr,
                                                           hkeys, hids, dimsCtr + 3, slot);
    k_assign<<<(n + TPB - 1) / TPB, TPB, 0, stream>>>(xyz, xyz_count, batch, n,
                                                      hids, slot, inv, counts, bsum, csum, xsum);
    k_ln_gemm_scatter<<<(n + 15) / 16, 128, 0, stream>>>(feat, ln_gamma, ln_beta, Wh,
                                                         inv, n, featU);
    k_finalize<<<4096, TPB, 0, stream>>>(counts, bsum, csum, xsum, n, out);
}
